// WindowAttention_45097156608399
// MI455X (gfx1250) — matrix operand order repaired and checked by reading, not compiled, not run on hardware
//
#include <hip/hip_runtime.h>
#include <hip/hip_bf16.h>
#include <math.h>
#include <stdint.h>

// ---------------------------------------------------------------------------
// Swin window attention for MI455X (gfx1250, wave32, WMMA + async-to-LDS).
//
//   [0] transpose+convert W_q/W_kv/W_proj to bf16 N-major (once, L2-resident)
//   [1] Q = (x @ W_q + b_q) * SCALE          -> ws_q  bf16 (b,h,tok,d)
//   [2] K,V = y @ W_kv + b_kv                -> ws_k (b,h,tok,d), ws_v (b,h,d,tok)
//   [3] per (b,h): softmax(QK^T + bias + mask) V -> ws_o bf16 (b,tok,h,d)
//   [4] out = ws_o @ W_proj + b_proj         -> d_out fp32
//
// GEMMs: block owns 64 rows x ALL columns; 64x512 bf16 A slab staged in LDS
// once (A read exactly once from HBM); W^T tiles double-buffered via
// global_load_async_to_lds_b128 with cross-iteration prefetch (the copy hides
// behind the previous tile's WMMAs, s_wait_asynccnt 2 synchronizes).
// All LDS operands laid out so every WMMA fragment is 2x ds_load_b128.
// ---------------------------------------------------------------------------

typedef __attribute__((ext_vector_type(16))) __bf16 v16bf;
typedef __attribute__((ext_vector_type(8)))  float  v8f;

#define DIM     512
#define NTOK    64
#define HEADS   16
#define HD      32
#define BATCH   2048
#define NWIN    64
#define MROWS   (BATCH * NTOK)        // 131072
#define SCALE_Q 0.17677669529663687f  // 32^-0.5

// ------------------------- CDNA5 async copy helpers -------------------------
__device__ inline void async_b128(unsigned lds_off, uint64_t gbase, unsigned gvoff) {
    asm volatile("global_load_async_to_lds_b128 %0, %1, %2"
                 :: "v"(lds_off), "v"(gvoff), "s"(gbase)
                 : "memory");
}
__device__ inline void wait_async0() {
    asm volatile("s_wait_asynccnt 0x0" ::: "memory");
}
__device__ inline void wait_async2() {   // leave the 2 newest prefetches in flight
    asm volatile("s_wait_asynccnt 0x2" ::: "memory");
}
__device__ inline unsigned lds_off_of(const void* p) {
    return (unsigned)(uintptr_t)p;   // low 32 bits of shared-aperture address
}

// ------------------------- ds_swizzle lane reductions -----------------------
template <int IMM>
__device__ inline float swz_f(float x) {
    return __builtin_bit_cast(
        float, __builtin_amdgcn_ds_swizzle(__builtin_bit_cast(int, x), IMM));
}
__device__ inline float redmax16(float v) {
    v = fmaxf(v, swz_f<0x041F>(v));
    v = fmaxf(v, swz_f<0x081F>(v));
    v = fmaxf(v, swz_f<0x101F>(v));
    v = fmaxf(v, swz_f<0x201F>(v));
    return v;
}
__device__ inline float redsum16(float v) {
    v += swz_f<0x041F>(v);
    v += swz_f<0x081F>(v);
    v += swz_f<0x101F>(v);
    v += swz_f<0x201F>(v);
    return v;
}

// -------------------------- WMMA fragment helpers ---------------------------
// A fragment (16x32 bf16): lane L: M=m0+(L&15), half=L>>4;
// a[2v+s] <- K = half*16 + 2v + s   -> one 32B contiguous run per lane
template <int LD>
__device__ inline v16bf load_frag_a(const __bf16* src, int m0, int k0, int lane) {
    const int mrow = m0 + (lane & 15);
    const int half = lane >> 4;
    v16bf a;
#pragma unroll
    for (int v = 0; v < 8; ++v) {
        const int kk = half * 16 + 2 * v;
        a[2 * v]     = src[mrow * LD + k0 + kk];
        a[2 * v + 1] = src[mrow * LD + k0 + kk + 1];
    }
    return a;
}
// B fragment from an N-major (transposed) operand: src row n holds K values.
// lane L: N=n0+(L&15), half=L>>4; b[2v+s] <- K = half*16 + 2v + s
// -> one 32B contiguous run per lane (2x ds_load_b128).
template <int LD>
__device__ inline v16bf load_frag_bt(const __bf16* src, int n0, int k0, int lane) {
    const int ncol = n0 + (lane & 15);
    const int half = lane >> 4;
    v16bf b;
#pragma unroll
    for (int v = 0; v < 8; ++v) {
        const int kk = half * 16 + 2 * v;
        b[2 * v]     = src[ncol * LD + k0 + kk];
        b[2 * v + 1] = src[ncol * LD + k0 + kk + 1];
    }
    return b;
}
__device__ inline v8f wmma_bf16(v16bf a, v16bf b, v8f c) {
    return __builtin_amdgcn_wmma_f32_16x16x32_bf16(false, a, false, b, (short)0, c,
                                                   false, false);
}

// --------------- weight fp32 -> bf16 transpose prepass (tiny) ---------------
// src: K x N row-major ; dst: N x K row-major (N-major for contiguous B frags)
__global__ void cvtT_kernel(const float* __restrict__ src, __bf16* __restrict__ dst,
                            int K, int N) {
    const int i = blockIdx.x * 256 + threadIdx.x;
    if (i < K * N) {
        const int k = i / N, n = i % N;
        dst[(size_t)n * K + k] = (__bf16)src[i];
    }
}

// ------------------------------- GEMM kernel --------------------------------
// Block: 256 threads (8 waves), 64 rows, all NCOLS columns.
// Wave w: M strip (w&3)*16, N strip (w>>2)*32 within the 64-col chunk.
enum { OUT_Q = 0, OUT_KV = 1, OUT_PROJ = 2 };

#define SA_LD 520   // 64 x 512 bf16 slab, 1040B/row (16B multiple)
#define SB_LD 72    // 64(n) x 64(k) bf16 tile, 144B/row (16B multiple)

template <int MODE, bool A_IS_BF16, int NCOLS>
__global__ __launch_bounds__(256, 2) void gemm_kernel(
    const void* __restrict__ Aptr, const __bf16* __restrict__ Wt,
    const float* __restrict__ bias, void* __restrict__ out0,
    void* __restrict__ out1, float scale) {
    __shared__ __bf16 sA[64 * SA_LD];
    __shared__ __bf16 sB[2][64 * SB_LD];

    const int tid  = threadIdx.x;
    const int lane = tid & 31;
    const int wave = tid >> 5;
    const int lm   = lane & 15;
    const int half = lane >> 4;
    const int wm   = (wave & 3) * 16;
    const int wn   = (wave >> 2) * 32;
    const int rowBase = blockIdx.x * 64;

    const uint64_t gW = (uint64_t)Wt;
    const unsigned sBb[2] = {lds_off_of(sB[0]), lds_off_of(sB[1])};

    // tile t: nc = (t/8)*64 columns, k0 = (t%8)*64 ; W^T tile rows = out cols
    auto issue_tile = [&](int t, int buf) {
        const int nc = (t >> 3) * 64, k0 = (t & 7) * 64;
#pragma unroll
        for (int i = 0; i < 2; ++i) {
            const int idx = (i * 256 + tid) * 8;   // 0..4088
            const int r = idx >> 6, c = idx & 63;  // r = col-in-chunk, c = k
            async_b128(sBb[buf] + (unsigned)(r * SB_LD + c) * 2, gW,
                       (unsigned)(((nc + r) * DIM + k0 + c) * 2));
        }
    };

    constexpr int T = (NCOLS / 64) * 8;
    issue_tile(0, 0);

    // ---- stage the 64x512 A slab once (overlaps with tile-0 async copy) ----
    if constexpr (A_IS_BF16) {
        const uint64_t gA = (uint64_t)Aptr + (uint64_t)rowBase * DIM * 2;
        const unsigned sAb = lds_off_of(sA);
#pragma unroll
        for (int i = 0; i < 16; ++i) {
            const int idx = (i * 256 + tid) * 8;   // 0..32760
            const int r = idx >> 9, c = idx & 511;
            async_b128(sAb + (unsigned)(r * SA_LD + c) * 2, gA, (unsigned)idx * 2);
        }
    } else {
        const float* Af = (const float*)Aptr;
#pragma unroll 4
        for (int i = 0; i < 128; ++i) {
            const int idx = tid + i * 256;
            const int r = idx >> 9, c = idx & 511;
            sA[r * SA_LD + c] = (__bf16)Af[(size_t)(rowBase + r) * DIM + c];
        }
    }

    v8f acc[2];
    for (int t = 0; t < T; ++t) {
        const int buf = t & 1;
        __syncthreads();                     // alternate buffer free for reuse
        if (t + 1 < T) {
            issue_tile(t + 1, buf ^ 1);      // prefetch next tile
            wait_async2();                   // tile t (and A slab) landed
        } else {
            wait_async0();
        }
        __syncthreads();                     // tile t visible to all waves

        if ((t & 7) == 0) {
#pragma unroll
            for (int j = 0; j < 2; ++j)
                acc[j] = (v8f){0.f, 0.f, 0.f, 0.f, 0.f, 0.f, 0.f, 0.f};
        }
        const int k0 = (t & 7) * 64;
        const __bf16* sBt = sB[buf];
#pragma unroll
        for (int kt = 0; kt < 2; ++kt) {
            const v16bf af = load_frag_a<SA_LD>(sA, wm, k0 + kt * 32, lane);
#pragma unroll
            for (int j = 0; j < 2; ++j) {
                const v16bf bfr = load_frag_bt<SB_LD>(sBt, wn + j * 16, kt * 32, lane);
                acc[j] = wmma_bf16(af, bfr, acc[j]);
            }
        }

        if ((t & 7) == 7) {   // ---- epilogue for this 64-col chunk ----
            const int nc = (t >> 3) * 64;
#pragma unroll
            for (int j = 0; j < 2; ++j) {
                const int colT = nc + wn + j * 16 + lm;
                const float bv = bias[colT];
#pragma unroll
                for (int r = 0; r < 8; ++r) {
                    const int rowT = rowBase + wm + r + 8 * half;
                    float val = acc[j][r] + bv;
                    if constexpr (MODE == OUT_PROJ) {
                        ((float*)out0)[(size_t)rowT * DIM + colT] = val;
                    } else {
                        val *= scale;
                        const int b = rowT >> 6, tok = rowT & 63;
                        if constexpr (MODE == OUT_Q) {
                            const int h = colT >> 5, d = colT & 31;
                            ((__bf16*)out0)[(((size_t)b * HEADS + h) * NTOK + tok) * HD + d] =
                                (__bf16)val;
                        } else {  // OUT_KV: [0,512)->K (tok-major), [512,1024)->V (d-major)
                            const int cc = colT & 511;
                            const int h = cc >> 5, d = cc & 31;
                            if (colT < 512)
                                ((__bf16*)out0)[(((size_t)b * HEADS + h) * NTOK + tok) * HD + d] =
                                    (__bf16)val;
                            else
                                ((__bf16*)out1)[(((size_t)b * HEADS + h) * HD + d) * NTOK + tok] =
                                    (__bf16)val;
                        }
                    }
                }
            }
        }
    }
}

// ----------------------------- Attention kernel -----------------------------
// One block (128 threads, 4 waves) per (b,h); wave w owns 16-token row strip.
#define SQ_LD 40   // q/k: 64 tok x 32 d, 80B/row
#define SV_LD 72   // v^T: 32 d x 64 tok, 144B/row
#define SP_LD 72   // probs: 64 x 64, 144B/row

__global__ __launch_bounds__(128, 4) void attn_kernel(
    const __bf16* __restrict__ q, const __bf16* __restrict__ k,
    const __bf16* __restrict__ v, const float* __restrict__ mask,
    const float* __restrict__ bias_table, __bf16* __restrict__ o) {
    __shared__ __bf16 sQ[NTOK * SQ_LD];
    __shared__ __bf16 sK[NTOK * SQ_LD];
    __shared__ __bf16 sVt[HD * SV_LD];     // (d, tok)
    __shared__ float  sBM[NTOK * NTOK];
    __shared__ __bf16 sP[NTOK * SP_LD];

    const int blk  = blockIdx.x;           // b*16 + h
    const int b    = blk >> 4;
    const int h    = blk & 15;
    const int tid  = threadIdx.x;
    const int lane = tid & 31;
    const int wave = tid >> 5;
    const int lm   = lane & 15;
    const int half = lane >> 4;
    const int m0   = wave * 16;

    // ---- async-stage q/k (tok-major) and v (d-major) tiles ----
    const uint64_t gq = (uint64_t)q + (uint64_t)blk * (NTOK * HD) * 2;
    const uint64_t gk = (uint64_t)k + (uint64_t)blk * (NTOK * HD) * 2;
    const uint64_t gv = (uint64_t)v + (uint64_t)blk * (NTOK * HD) * 2;
    const unsigned sQb = lds_off_of(sQ), sKb = lds_off_of(sK), sVb = lds_off_of(sVt);
#pragma unroll
    for (int i = 0; i < 2; ++i) {
        const int idx = (i * 128 + tid) * 8;   // 0..2040
        const unsigned go = (unsigned)idx * 2;
        {   // q/k: r = tok, c = d
            const int r = idx >> 5, c = idx & 31;
            const unsigned lo = (unsigned)(r * SQ_LD + c) * 2;
            async_b128(sQb + lo, gq, go);
            async_b128(sKb + lo, gk, go);
        }
        {   // v^T: r = d, c = tok
            const int r = idx >> 6, c = idx & 63;
            async_b128(sVb + (unsigned)(r * SV_LD + c) * 2, gv, go);
        }
    }
    // ---- stage bias[REL_IDX(m,n)] + mask ----
    const float* mrow = mask + (size_t)(b & (NWIN - 1)) * (NTOK * NTOK);
#pragma unroll
    for (int i = 0; i < 32; ++i) {
        const int idx = tid + i * 128;         // 0..4095
        const int m = idx >> 6, n = idx & 63;
        const int rel = ((m >> 3) - (n >> 3) + 7) * 15 + ((m & 7) - (n & 7) + 7);
        sBM[idx] = bias_table[rel * HEADS + h] + mrow[idx];
    }
    wait_async0();
    __syncthreads();

    // ---- S = q @ k^T (q pre-scaled): 4 WMMA per wave ----
    const v16bf aq = load_frag_a<SQ_LD>(sQ, m0, 0, lane);
    v8f s_acc[4];
#pragma unroll
    for (int nt = 0; nt < 4; ++nt) {
        const v16bf bk = load_frag_bt<SQ_LD>(sK, nt * 16, 0, lane);
        v8f z = (v8f){0.f, 0.f, 0.f, 0.f, 0.f, 0.f, 0.f, 0.f};
        s_acc[nt] = wmma_bf16(aq, bk, z);
    }

    // ---- + bias + mask, softmax (row m lives in one 16-lane group) ----
#pragma unroll
    for (int nt = 0; nt < 4; ++nt)
#pragma unroll
        for (int r = 0; r < 8; ++r)
            s_acc[nt][r] += sBM[(m0 + r + 8 * half) * NTOK + nt * 16 + lm];

#pragma unroll
    for (int r = 0; r < 8; ++r) {
        float mx = fmaxf(fmaxf(s_acc[0][r], s_acc[1][r]),
                         fmaxf(s_acc[2][r], s_acc[3][r]));
        mx = redmax16(mx);
        float sum = 0.f;
#pragma unroll
        for (int nt = 0; nt < 4; ++nt) {
            const float e = __expf(s_acc[nt][r] - mx);
            s_acc[nt][r] = e;
            sum += e;
        }
        sum = redsum16(sum);
        const float inv = 1.0f / sum;
#pragma unroll
        for (int nt = 0; nt < 4; ++nt)
            sP[(m0 + r + 8 * half) * SP_LD + nt * 16 + lm] =
                (__bf16)(s_acc[nt][r] * inv);
    }
    __syncthreads();

    // ---- O = P @ V (K=64 -> 2 chunks): 4 WMMA per wave ----
    v16bf ap[2];
#pragma unroll
    for (int kt = 0; kt < 2; ++kt) ap[kt] = load_frag_a<SP_LD>(sP, m0, kt * 32, lane);

    v8f oacc[2];
#pragma unroll
    for (int nt = 0; nt < 2; ++nt)
        oacc[nt] = (v8f){0.f, 0.f, 0.f, 0.f, 0.f, 0.f, 0.f, 0.f};
#pragma unroll
    for (int nt = 0; nt < 2; ++nt)
#pragma unroll
        for (int kt = 0; kt < 2; ++kt)
            oacc[nt] = wmma_bf16(ap[kt],
                                 load_frag_bt<SV_LD>(sVt, nt * 16, kt * 32, lane),
                                 oacc[nt]);

    // ---- store O as (b, tok, h, d) bf16 ----
#pragma unroll
    for (int nt = 0; nt < 2; ++nt) {
        const int d = nt * 16 + lm;
#pragma unroll
        for (int r = 0; r < 8; ++r) {
            const int mrel = m0 + r + 8 * half;
            o[((size_t)b * NTOK + mrel) * DIM + h * HD + d] = (__bf16)oacc[nt][r];
        }
    }
}

// -------------------------------- launcher ----------------------------------
extern "C" void kernel_launch(void* const* d_in, const int* in_sizes, int n_in,
                              void* d_out, int out_size, void* d_ws, size_t ws_size,
                              hipStream_t stream) {
    const float* x          = (const float*)d_in[0];
    const float* y          = (const float*)d_in[1];
    const float* mask       = (const float*)d_in[2];
    const float* W_q        = (const float*)d_in[3];
    const float* b_q        = (const float*)d_in[4];
    const float* W_kv       = (const float*)d_in[5];
    const float* b_kv       = (const float*)d_in[6];
    const float* W_proj     = (const float*)d_in[7];
    const float* b_proj     = (const float*)d_in[8];
    const float* bias_table = (const float*)d_in[9];

    const size_t TEN = (size_t)MROWS * DIM * sizeof(__bf16);  // 128 MiB
    char* ws = (char*)d_ws;
    __bf16* ws_q     = (__bf16*)(ws);
    __bf16* ws_k     = (__bf16*)(ws + TEN);
    __bf16* ws_v     = (__bf16*)(ws + 2 * TEN);
    __bf16* ws_o     = (__bf16*)(ws + 3 * TEN);
    __bf16* ws_wq    = (__bf16*)(ws + 4 * TEN);
    __bf16* ws_wkv   = (__bf16*)(ws + 4 * TEN + (size_t)DIM * DIM * 2);
    __bf16* ws_wproj = (__bf16*)(ws + 4 * TEN + (size_t)DIM * DIM * 2 * 3);

    // [0] weight transpose+convert prepass (W^T, bf16, N-major)
    cvtT_kernel<<<dim3((DIM * DIM + 255) / 256), dim3(256), 0, stream>>>(
        W_q, ws_wq, DIM, DIM);
    cvtT_kernel<<<dim3((2 * DIM * DIM + 255) / 256), dim3(256), 0, stream>>>(
        W_kv, ws_wkv, DIM, 2 * DIM);
    cvtT_kernel<<<dim3((DIM * DIM + 255) / 256), dim3(256), 0, stream>>>(
        W_proj, ws_wproj, DIM, DIM);

    // [1] Q projection, [2] KV projection, [3] attention, [4] out projection
    gemm_kernel<OUT_Q, false, DIM><<<dim3(MROWS / 64), dim3(256), 0, stream>>>(
        x, ws_wq, b_q, ws_q, nullptr, SCALE_Q);
    gemm_kernel<OUT_KV, false, 2 * DIM><<<dim3(MROWS / 64), dim3(256), 0, stream>>>(
        y, ws_wkv, b_kv, ws_k, ws_v, 1.0f);
    attn_kernel<<<dim3(BATCH * HEADS), dim3(128), 0, stream>>>(
        ws_q, ws_k, ws_v, mask, bias_table, ws_o);
    gemm_kernel<OUT_PROJ, true, DIM><<<dim3(MROWS / 64), dim3(256), 0, stream>>>(
        ws_o, ws_wproj, b_proj, d_out, nullptr, 1.0f);
}
